// TemporalAttentionDecoder_14774687498322
// MI455X (gfx1250) — matrix operand order repaired and checked by reading, not compiled, not run on hardware
//
#include <hip/hip_runtime.h>

// Problem constants (from reference): B=2048, T=128, M=256, P=256
#define B_SZ 2048
#define T_SZ 128
#define M_SZ 256
#define P_SZ 256
#define NB 2          // batch rows owned per workgroup
#define NTHREADS 256  // 8 waves (wave32)

typedef __attribute__((ext_vector_type(16))) __bf16 v16bf;
typedef __attribute__((ext_vector_type(8)))  float  v8f;

union BFU { unsigned short u; __bf16 b; };

static __device__ __forceinline__ __bf16 bf_from_bits(unsigned short u) {
  BFU x; x.u = u; return x.b;
}
static __device__ __forceinline__ unsigned short f2bf(float f) {
  unsigned u = __float_as_uint(f);
  u += 0x7FFFu + ((u >> 16) & 1u);       // round-to-nearest-even
  return (unsigned short)(u >> 16);
}
static __device__ __forceinline__ float bf2f(unsigned short h) {
  return __uint_as_float(((unsigned)h) << 16);
}
static __device__ __forceinline__ float sigm(float x) {
  return 1.0f / (1.0f + __expf(-x));
}

// ws layout (bf16 elements): Ud [0,65536) | Wd [65536,196608) | Whh [196608,458752)
#define WS_UD  0
#define WS_WD  65536
#define WS_WHH 196608

__global__ void cvt_weights_kernel(const float* __restrict__ Ud,
                                   const float* __restrict__ Wd,
                                   const float* __restrict__ Whh,
                                   unsigned short* __restrict__ ws) {
  const int i = blockIdx.x * blockDim.x + threadIdx.x;  // grid = 262144 threads
  if (i < 65536)  ws[WS_UD + i]  = f2bf(Ud[i]);         // 256x256
  if (i < 131072) ws[WS_WD + i]  = f2bf(Wd[i]);         // 256x512
  ws[WS_WHH + i] = f2bf(Whh[i]);                        // 1024x256
}

__global__ __launch_bounds__(NTHREADS, 1)
void decoder_kernel(const float* __restrict__ enc,
                    const float* __restrict__ Wd_b,
                    const float* __restrict__ vd_w,
                    const float* __restrict__ wt_w,
                    const float* __restrict__ wt_b,
                    const float* __restrict__ Wih,
                    const float* __restrict__ bih,
                    const float* __restrict__ bhh,
                    const float* __restrict__ Wy_w,
                    const float* __restrict__ Wy_b,
                    const float* __restrict__ vy_w,
                    const float* __restrict__ vy_b,
                    const unsigned short* __restrict__ wsb,
                    float* __restrict__ out) {
  extern __shared__ char smem[];
  unsigned short* encL = (unsigned short*)smem;          // NB*T*M bf16 = 65536
  unsigned short* y1L  = encL + NB * T_SZ * M_SZ;        // NB*T*M bf16 = 65536
  float* fb   = (float*)(y1L + NB * T_SZ * M_SZ);
  float* dL   = fb;            // [NB][256]
  float* sL   = dL  + 512;     // [NB][256]
  float* x1L  = sL  + 512;     // [NB][256]
  float* cL   = x1L + 512;     // [NB][256]
  float* lL   = cL  + 512;     // [NB][128] scores -> beta
  float* red  = lL  + 256;     // [512] reduction scratch
  float* vdL  = red + 512;     // [256]
  float* wtL  = vdL + 256;     // [256]
  float* bbL  = wtL + 256;     // [1024] bih+bhh
  float* wihL = bbL + 1024;    // [1024]
  float* ytL  = wihL + 1024;   // [2]

  const int tid = threadIdx.x;
  const int b0  = blockIdx.x * NB;
  const float wtb = wt_b[0];
  const float vyb = vy_b[0];

  // ---- init small LDS constants + state ----
  vdL[tid] = vd_w[tid];
  wtL[tid] = wt_w[tid];
#pragma unroll
  for (int j = 0; j < 4; ++j) {
    bbL[j * 256 + tid]  = bih[j * 256 + tid] + bhh[j * 256 + tid];
    wihL[j * 256 + tid] = Wih[j * 256 + tid];
  }
  dL[tid] = 0.f; dL[256 + tid] = 0.f;
  sL[tid] = 0.f; sL[256 + tid] = 0.f;

  // ---- load this WG's enc rows into LDS as bf16 ----
  const float* encG = enc + (size_t)b0 * T_SZ * M_SZ;
  for (int i = tid; i < NB * T_SZ * M_SZ; i += NTHREADS) encL[i] = f2bf(encG[i]);
  __syncthreads();

  // ---- y1 = enc @ Ud^T via WMMA bf16 (f32 accumulate), result -> LDS bf16 ----
  {
    const int lane   = tid & 31;
    const int waveId = tid >> 5;
    const unsigned short* UdB = wsb + WS_UD;  // [256][256] row-major (n,k)
    for (int rtt = waveId; rtt < 16; rtt += 8) {   // 2 (r, t-tile) pairs per wave
      const int r  = rtt >> 3;
      const int tt = rtt & 7;
      const unsigned short* aBase = encL + (r * T_SZ + tt * 16) * M_SZ;
      // A fragments for all 8 K-steps (16x32 bf16 tile layout, ISA 7.12.2)
      v16bf afrag[8];
      const int arow  = lane & 15;
      const int khalf = (lane >> 4) << 3;    // 0 or 8
#pragma unroll
      for (int kt = 0; kt < 8; ++kt) {
        const unsigned short* p = aBase + arow * M_SZ + kt * 32;
#pragma unroll
        for (int j = 0; j < 8; ++j) afrag[kt][j]     = bf_from_bits(p[khalf + j]);
#pragma unroll
        for (int j = 0; j < 8; ++j) afrag[kt][8 + j] = bf_from_bits(p[16 + khalf + j]);
      }
      const int n    = lane & 15;
      const int koff = (lane >> 4) << 3;     // 0 or 8 (B: same K order as A: 8h+j, 16+8h+j)
      for (int nt = 0; nt < 16; ++nt) {
        v8f acc = {};
#pragma unroll
        for (int kt = 0; kt < 8; ++kt) {
          const unsigned short* q = UdB + (nt * 16 + n) * M_SZ + kt * 32 + koff;
          v16bf bfrag;
#pragma unroll
          for (int j = 0; j < 16; ++j) bfrag[j] = bf_from_bits(q[j < 8 ? j : 8 + j]);
          acc = __builtin_amdgcn_wmma_f32_16x16x32_bf16(
              false, afrag[kt], false, bfrag, (short)0, acc, false, false);
        }
        // C/D layout: lanes 0-15 -> rows 0..7 (vgpr i = row i), lanes 16-31 -> rows 8..15
        const int mb = (lane >> 4) << 3;
        unsigned short* dst = y1L + (r * T_SZ + tt * 16 + mb) * M_SZ + nt * 16 + n;
#pragma unroll
        for (int i = 0; i < 8; ++i) dst[i * M_SZ] = f2bf(acc[i]);
      }
    }
  }
  __syncthreads();

  const unsigned* WdB32  = (const unsigned*)(wsb + WS_WD);   // [256][512] bf16 -> 256 uint/row
  const unsigned* WhhB32 = (const unsigned*)(wsb + WS_WHH);  // [1024][256] bf16 -> 128 uint/row

  // ================== scan over 128 LSTM steps, all from LDS ==================
  for (int step = 0; step < T_SZ; ++step) {
    // (a) x1[r][m] = [d,s] . Wd_w[m,:] + Wd_b[m]   (m = tid)
    {
      const float wb = Wd_b[tid];
      const unsigned* wr = WdB32 + tid * 256;
#pragma unroll 1
      for (int r = 0; r < NB; ++r) {
        const float* dr = dL + r * 256;
        const float* sr = sL + r * 256;
        float acc = wb;
#pragma unroll 4
        for (int k2 = 0; k2 < 128; ++k2) {
          const unsigned w = wr[k2];
          acc = fmaf(bf2f((unsigned short)w),         dr[2 * k2],     acc);
          acc = fmaf(bf2f((unsigned short)(w >> 16)), dr[2 * k2 + 1], acc);
        }
#pragma unroll 4
        for (int k2 = 0; k2 < 128; ++k2) {
          const unsigned w = wr[128 + k2];
          acc = fmaf(bf2f((unsigned short)w),         sr[2 * k2],     acc);
          acc = fmaf(bf2f((unsigned short)(w >> 16)), sr[2 * k2 + 1], acc);
        }
        x1L[r * 256 + tid] = acc;
      }
    }
    __syncthreads();

    // (b) l[r][t] = sum_m tanh(x1 + y1) * vd   (one (r,t) pair per thread)
    {
      const int r = tid >> 7, t = tid & 127;
      const unsigned* yrow = (const unsigned*)y1L + ((r * T_SZ + t) * M_SZ >> 1);
      const float* xr = x1L + r * 256;
      float acc = 0.f;
#pragma unroll 4
      for (int m2 = 0; m2 < 128; ++m2) {
        const unsigned yv = yrow[m2];
        const float z0 = tanhf(xr[2 * m2]     + bf2f((unsigned short)yv));
        const float z1 = tanhf(xr[2 * m2 + 1] + bf2f((unsigned short)(yv >> 16)));
        acc = fmaf(z0, vdL[2 * m2], fmaf(z1, vdL[2 * m2 + 1], acc));
      }
      lL[tid] = acc;
    }
    __syncthreads();

    // (c) softmax over t (two 128-wide rows in parallel)
    {
      const int r = tid >> 7, tl = tid & 127;
      red[tid] = lL[tid];
      __syncthreads();
      for (int off = 64; off > 0; off >>= 1) {
        if (tl < off) red[tid] = fmaxf(red[tid], red[tid + off]);
        __syncthreads();
      }
      const float mx = red[r << 7];
      __syncthreads();
      const float e = __expf(lL[tid] - mx);
      red[tid] = e;
      __syncthreads();
      for (int off = 64; off > 0; off >>= 1) {
        if (tl < off) red[tid] += red[tid + off];
        __syncthreads();
      }
      const float ssum = red[r << 7];
      __syncthreads();
      lL[tid] = e / ssum;   // beta
    }
    __syncthreads();

    // (d) c[r][m] = sum_t beta[r][t] * enc[r][t][m]   (m = tid)
#pragma unroll 1
    for (int r = 0; r < NB; ++r) {
      const unsigned short* ecol = encL + r * T_SZ * M_SZ + tid;
      const float* br = lL + r * 128;
      float acc = 0.f;
#pragma unroll 4
      for (int t = 0; t < T_SZ; ++t) acc = fmaf(br[t], bf2f(ecol[t * M_SZ]), acc);
      cL[r * 256 + tid] = acc;
    }
    __syncthreads();

    // (e) y_t[r] = c[r] . wt_w + wt_b   (two tree reductions)
    red[tid]       = cL[tid]       * wtL[tid];
    red[256 + tid] = cL[256 + tid] * wtL[tid];
    __syncthreads();
    for (int off = 128; off > 0; off >>= 1) {
      if (tid < off) {
        red[tid]       += red[tid + off];
        red[256 + tid] += red[256 + tid + off];
      }
      __syncthreads();
    }
    if (tid == 0) { ytL[0] = red[0] + wtb; ytL[1] = red[256] + wtb; }
    __syncthreads();

    // (f) gates + LSTM cell update (hidden unit p = tid)
    {
      float dn[NB], sn[NB];
#pragma unroll 1
      for (int r = 0; r < NB; ++r) {
        const float yt = ytL[r];
        const float* dr = dL + r * 256;
        float gv[4];
#pragma unroll 1
        for (int gate = 0; gate < 4; ++gate) {
          const int j = gate * 256 + tid;
          float acc = fmaf(yt, wihL[j], bbL[j]);
          const unsigned* wr = WhhB32 + j * 128;
#pragma unroll 4
          for (int k2 = 0; k2 < 128; ++k2) {
            const unsigned w = wr[k2];
            acc = fmaf(bf2f((unsigned short)w),         dr[2 * k2],     acc);
            acc = fmaf(bf2f((unsigned short)(w >> 16)), dr[2 * k2 + 1], acc);
          }
          gv[gate] = acc;
        }
        const float ig = sigm(gv[0]), fg = sigm(gv[1]);
        const float gg = tanhf(gv[2]), og = sigm(gv[3]);
        const float s_new = fg * sL[r * 256 + tid] + ig * gg;
        sn[r] = s_new;
        dn[r] = og * tanhf(s_new);
      }
      __syncthreads();
#pragma unroll
      for (int r = 0; r < NB; ++r) {
        dL[r * 256 + tid] = dn[r];
        sL[r * 256 + tid] = sn[r];
      }
    }
    __syncthreads();
  }

  // ---- final head: h = [d,c] @ Wy^T + Wy_b ; out = h . vy_w + vy_b ----
  {
    float p[NB];
    const float wyb = Wy_b[tid];
    const float vyw = vy_w[tid];
    const float* wr = Wy_w + tid * 512;
#pragma unroll 1
    for (int r = 0; r < NB; ++r) {
      const float* dr = dL + r * 256;
      const float* cr = cL + r * 256;
      float acc = wyb;
#pragma unroll 4
      for (int k = 0; k < 256; ++k) acc = fmaf(wr[k], dr[k], acc);
#pragma unroll 4
      for (int k = 0; k < 256; ++k) acc = fmaf(wr[256 + k], cr[k], acc);
      p[r] = acc * vyw;
    }
    red[tid] = p[0];
    red[256 + tid] = p[1];
    __syncthreads();
    for (int off = 128; off > 0; off >>= 1) {
      if (tid < off) {
        red[tid]       += red[tid + off];
        red[256 + tid] += red[256 + tid + off];
      }
      __syncthreads();
    }
    if (tid == 0) {
      out[b0]     = red[0]   + vyb;
      out[b0 + 1] = red[256] + vyb;
    }
  }
}

extern "C" void kernel_launch(void* const* d_in, const int* in_sizes, int n_in,
                              void* d_out, int out_size, void* d_ws, size_t ws_size,
                              hipStream_t stream) {
  (void)in_sizes; (void)n_in; (void)out_size; (void)ws_size;
  const float* enc  = (const float*)d_in[0];
  const float* Wd_w = (const float*)d_in[1];
  const float* Wd_b = (const float*)d_in[2];
  const float* Ud_w = (const float*)d_in[3];
  const float* vd_w = (const float*)d_in[4];
  const float* wt_w = (const float*)d_in[5];
  const float* wt_b = (const float*)d_in[6];
  const float* Wih  = (const float*)d_in[7];
  const float* Whh  = (const float*)d_in[8];
  const float* bih  = (const float*)d_in[9];
  const float* bhh  = (const float*)d_in[10];
  const float* Wy_w = (const float*)d_in[11];
  const float* Wy_b = (const float*)d_in[12];
  const float* vy_w = (const float*)d_in[13];
  const float* vy_b = (const float*)d_in[14];
  unsigned short* wsb = (unsigned short*)d_ws;
  float* out = (float*)d_out;

  // bf16 weight staging: 262144 threads cover Ud(64K)/Wd(128K)/Whh(256K)
  cvt_weights_kernel<<<1024, 256, 0, stream>>>(Ud_w, Wd_w, Whh, wsb);

  // enc(bf16) + y1(bf16) tiles for 2 batch rows + f32 state/scratch: ~284 KB LDS (<= 320 KB/WGP)
  constexpr size_t LDS_BYTES =
      (size_t)(2 * NB * T_SZ * M_SZ) * 2 /*enc+y1 bf16*/ + 5378u * 4 /*f32 pools*/ + 64;
  decoder_kernel<<<B_SZ / NB, NTHREADS, LDS_BYTES, stream>>>(
      enc, Wd_b, vd_w, wt_w, wt_b, Wih, bih, bhh, Wy_w, Wy_b, vy_w, vy_b, wsb, out);
}
